// Net_11416023073012
// MI455X (gfx1250) — compile-verified, matrix operand order checked by reading, not run on hardware
//
#include <hip/hip_runtime.h>
#include <math.h>

#define N_NODES 100000
#define N_EDGES 3200000
#define F_IN    512
#define HID     16
#define NCLS    10
#define NTILES  (N_NODES / 16)   // 6250, exact

typedef __attribute__((ext_vector_type(16))) __bf16 v16bf;
typedef __attribute__((ext_vector_type(8)))  float  v8f;

// ---------------- degree / norm pipeline ----------------

__global__ void k_init_deg(float* __restrict__ deg) {
    int i = blockIdx.x * blockDim.x + threadIdx.x;
    if (i < N_NODES) deg[i] = 1.0f;   // self-loop weight 1
}

__global__ void k_deg_edges(const int* __restrict__ dst, const float* __restrict__ w,
                            float* __restrict__ deg) {
    int e = blockIdx.x * blockDim.x + threadIdx.x;
    if (e < N_EDGES) atomicAdd(&deg[dst[e]], w[e]);
}

__global__ void k_dinv(float* __restrict__ deg) {
    int i = blockIdx.x * blockDim.x + threadIdx.x;
    if (i < N_NODES) {
        float d = deg[i];
        deg[i] = (d > 0.0f) ? rsqrtf(d) : 0.0f;
    }
}

__global__ void k_norm(const int* __restrict__ src, const int* __restrict__ dst,
                       const float* __restrict__ w, const float* __restrict__ dinv,
                       float* __restrict__ nrm) {
    int e = blockIdx.x * blockDim.x + threadIdx.x;
    if (e < N_EDGES) nrm[e] = dinv[src[e]] * w[e] * dinv[dst[e]];
}

// ---------------- WMMA GEMM: h = x @ W1  ([N,512]x[512,16]) ----------------
// One wave per 16-row tile. W1 cached in LDS as bf16.
// A (16x32 bf16) per-lane layout: row = lane&15; half h = lane>>4;
//   vgpr pair j<4: K = k0 + 2j + 8h ; j>=4: K = k0 + 16 + 2(j-4) + 8h.
// B (32x16) mirrors with col = lane&15.  D (16x16 f32): vgpr r -> row r + 8h, col lane&15.

__global__ void k_gemm1(const float* __restrict__ x, const float* __restrict__ W1,
                        float* __restrict__ h) {
    __shared__ __bf16 w1s[F_IN * HID];           // 16 KB
    int t = threadIdx.x;
    for (int i = t; i < F_IN * HID; i += blockDim.x) w1s[i] = (__bf16)W1[i];
    __syncthreads();

    int wave = t >> 5;
    int lane = t & 31;
    int tile = blockIdx.x * 8 + wave;
    if (tile >= NTILES) return;

    int col = lane & 15;
    int hh  = lane >> 4;
    const float* xr = x + (size_t)(tile * 16 + col) * F_IN;  // A row for this lane

    v8f acc;
    #pragma unroll
    for (int r = 0; r < 8; ++r) acc[r] = 0.0f;

    for (int k0 = 0; k0 < F_IN; k0 += 32) {
        v16bf a, b;
        #pragma unroll
        for (int j = 0; j < 8; ++j) {
            int kl = (j < 4) ? (2 * j + 8 * hh) : (16 + 2 * (j - 4) + 8 * hh);
            int k  = k0 + kl;
            float2 xv = *reinterpret_cast<const float2*>(xr + k);
            a[2 * j]     = (__bf16)xv.x;
            a[2 * j + 1] = (__bf16)xv.y;
            b[2 * j]     = w1s[k * HID + col];
            b[2 * j + 1] = w1s[(k + 1) * HID + col];
        }
        acc = __builtin_amdgcn_wmma_f32_16x16x32_bf16(
                  false, a, false, b, (short)0, acc, false, false);
    }

    float* out = h + (size_t)(tile * 16 + 8 * hh) * HID + col;
    #pragma unroll
    for (int r = 0; r < 8; ++r) out[(size_t)r * HID] = acc[r];
}

// ---------------- WMMA GEMM: h2 = h1 @ W2  ([N,16]x[16,10], K/N zero-padded) ----

__global__ void k_gemm2(const float* __restrict__ h1, const float* __restrict__ W2,
                        float* __restrict__ h2) {
    __shared__ __bf16 w2s[16 * 16];
    int t = threadIdx.x;
    for (int i = t; i < 16 * 16; i += blockDim.x) {
        int k = i >> 4, n = i & 15;
        w2s[i] = (n < NCLS) ? (__bf16)W2[k * NCLS + n] : (__bf16)0.0f;
    }
    __syncthreads();

    int wave = t >> 5;
    int lane = t & 31;
    int tile = blockIdx.x * 8 + wave;
    if (tile >= NTILES) return;

    int col = lane & 15;
    int hh  = lane >> 4;
    const float* hr = h1 + (size_t)(tile * 16 + col) * HID;

    v16bf a, b;
    #pragma unroll
    for (int i = 0; i < 16; ++i) { a[i] = (__bf16)0.0f; b[i] = (__bf16)0.0f; }
    #pragma unroll
    for (int j = 0; j < 8; ++j) {
        int kl = (j < 4) ? (2 * j + 8 * hh) : (16 + 2 * (j - 4) + 8 * hh);
        if (kl < 16) {                       // K padded 16 -> 32 with zeros
            a[2 * j]     = (__bf16)hr[kl];
            a[2 * j + 1] = (__bf16)hr[kl + 1];
            b[2 * j]     = w2s[kl * 16 + col];
            b[2 * j + 1] = w2s[(kl + 1) * 16 + col];
        }
    }
    v8f acc;
    #pragma unroll
    for (int r = 0; r < 8; ++r) acc[r] = 0.0f;
    acc = __builtin_amdgcn_wmma_f32_16x16x32_bf16(
              false, a, false, b, (short)0, acc, false, false);

    float* out = h2 + (size_t)(tile * 16 + 8 * hh) * HID + col;
    #pragma unroll
    for (int r = 0; r < 8; ++r) out[(size_t)r * HID] = acc[r];
}

// ---------------- aggregation (self-loop init + edge scatter-add) ----------------

__global__ void k_self_init(const float* __restrict__ dinv, const float* __restrict__ h,
                            float* __restrict__ agg) {
    int i = blockIdx.x * blockDim.x + threadIdx.x;
    if (i < N_NODES * HID) {
        float di = dinv[i >> 4];
        agg[i] = di * di * h[i];             // self-loop: norm = dinv^2 * w(=1)
    }
}

__global__ void k_scatter(const int* __restrict__ src, const int* __restrict__ dst,
                          const float* __restrict__ nrm, const float* __restrict__ h,
                          float* __restrict__ agg) {
    long long i = (long long)blockIdx.x * blockDim.x + threadIdx.x;
    if (i >= (long long)N_EDGES * HID) return;
    int e = (int)(i >> 4);
    int c = (int)(i & 15);
    float nv = nrm[e];                                        // broadcast across 16 lanes
    atomicAdd(&agg[(size_t)dst[e] * HID + c], nv * h[(size_t)src[e] * HID + c]);
}

__global__ void k_bias_relu(const float* __restrict__ b1, float* __restrict__ agg) {
    int i = blockIdx.x * blockDim.x + threadIdx.x;
    if (i < N_NODES * HID) {
        float v = agg[i] + b1[i & 15];
        agg[i] = v > 0.0f ? v : 0.0f;
    }
}

// ---------------- bias + log_softmax over 10 classes ----------------

__global__ void k_logsoftmax(const float* __restrict__ agg2, const float* __restrict__ b2,
                             float* __restrict__ out) {
    int i = blockIdx.x * blockDim.x + threadIdx.x;
    if (i >= N_NODES) return;
    float z[NCLS];
    float m = -INFINITY;
    #pragma unroll
    for (int c = 0; c < NCLS; ++c) {
        z[c] = agg2[(size_t)i * HID + c] + b2[c];
        m = fmaxf(m, z[c]);
    }
    float s = 0.0f;
    #pragma unroll
    for (int c = 0; c < NCLS; ++c) s += __expf(z[c] - m);
    float ls = __logf(s) + m;
    #pragma unroll
    for (int c = 0; c < NCLS; ++c) out[(size_t)i * NCLS + c] = z[c] - ls;
}

// ---------------- launcher ----------------

extern "C" void kernel_launch(void* const* d_in, const int* in_sizes, int n_in,
                              void* d_out, int out_size, void* d_ws, size_t ws_size,
                              hipStream_t stream) {
    const float* x  = (const float*)d_in[0];
    const int*   ei = (const int*)  d_in[1];     // [2, E] flat: row0=src, row1=dst
    const float* ew = (const float*)d_in[2];
    const float* W1 = (const float*)d_in[3];
    const float* b1 = (const float*)d_in[4];
    const float* W2 = (const float*)d_in[5];
    const float* b2 = (const float*)d_in[6];
    float* out = (float*)d_out;

    const int* srcp = ei;
    const int* dstp = ei + N_EDGES;

    // workspace layout (floats): ~38.8 MB total
    float* ws   = (float*)d_ws;
    float* dinv = ws;                                  // N       (deg -> dinv in place)
    float* nrm  = dinv + N_NODES;                      // E
    float* hbuf = nrm  + N_EDGES;                      // N*16
    float* agg1 = hbuf + (size_t)N_NODES * HID;        // N*16    (becomes h1 after relu)
    float* h2   = agg1 + (size_t)N_NODES * HID;        // N*16    (cols 10..15 are zero)
    float* agg2 = h2   + (size_t)N_NODES * HID;        // N*16

    const int B = 256;
    int gN   = (N_NODES + B - 1) / B;
    int gE   = (N_EDGES + B - 1) / B;
    int gNH  = (N_NODES * HID + B - 1) / B;
    long long eh = (long long)N_EDGES * HID;
    int gEH  = (int)((eh + B - 1) / B);
    int gMM  = (NTILES + 7) / 8;                       // 8 waves (16-row tiles) per block

    // normalization coefficients
    k_init_deg <<<gN, B, 0, stream>>>(dinv);
    k_deg_edges<<<gE, B, 0, stream>>>(dstp, ew, dinv);
    k_dinv     <<<gN, B, 0, stream>>>(dinv);
    k_norm     <<<gE, B, 0, stream>>>(srcp, dstp, ew, dinv, nrm);

    // layer 1: h = x @ W1 ; agg1 = D^-1/2 A_hat D^-1/2 h ; h1 = relu(agg1 + b1)
    k_gemm1    <<<gMM, B, 0, stream>>>(x, W1, hbuf);
    k_self_init<<<gNH, B, 0, stream>>>(dinv, hbuf, agg1);
    k_scatter  <<<gEH, B, 0, stream>>>(srcp, dstp, nrm, hbuf, agg1);
    k_bias_relu<<<gNH, B, 0, stream>>>(b1, agg1);

    // layer 2: h2 = h1 @ W2 ; agg2 = aggregate ; out = log_softmax(agg2 + b2)
    k_gemm2    <<<gMM, B, 0, stream>>>(agg1, W2, h2);
    k_self_init<<<gNH, B, 0, stream>>>(dinv, h2, agg2);
    k_scatter  <<<gEH, B, 0, stream>>>(srcp, dstp, nrm, h2, agg2);
    k_logsoftmax<<<gN, B, 0, stream>>>(agg2, b2, out);
}
